// LearnedSimulator_12704513261570
// MI455X (gfx1250) — compile-verified, matrix operand order checked by reading, not run on hardware
//
#include <hip/hip_runtime.h>
#include <hip/hip_bf16.h>

// ---------------------------------------------------------------------------
// GNS learned-simulator forward on gfx1250 (CDNA5, wave32, WMMA bf16).
// All 128-wide GEMMs run on v_wmma_f32_16x16x32_bf16; LDS staging uses
// GLOBAL_LOAD_ASYNC_TO_LDS_B128 when the toolchain exposes the builtin.
// ---------------------------------------------------------------------------

typedef __bf16 bf16;
typedef __attribute__((ext_vector_type(16))) bf16  v16bf;
typedef __attribute__((ext_vector_type(8)))  bf16  v8bf;
typedef __attribute__((ext_vector_type(8)))  float v8f;
typedef __attribute__((ext_vector_type(4)))  int   v4i;

#define HID 128

#if defined(__has_builtin)
#if __has_builtin(__builtin_amdgcn_global_load_async_to_lds_b128) && \
    __has_builtin(__builtin_amdgcn_s_wait_asynccnt)
#define HAVE_ASYNC_LDS 1
#endif
#endif
#ifndef HAVE_ASYNC_LDS
#define HAVE_ASYNC_LDS 0
#endif

#if HAVE_ASYNC_LDS
// per-lane async copy of 16 bytes global->LDS (tracked by ASYNCcnt)
#define ASYNC_B128(gsrc, ldst)                                              \
    __builtin_amdgcn_global_load_async_to_lds_b128(                         \
        (__attribute__((address_space(1))) v4i*)(void*)(gsrc),              \
        (__attribute__((address_space(3))) v4i*)(void*)(ldst), 0, 0)
#endif

// K index inside a 16x32 bf16 A/B fragment: lane supplies khalf = lane>>4,
// element j in 0..15.  (CDNA5 ISA, 16-bit A-matrix layout)
__device__ __host__ __forceinline__ int wmma_k_of(int j, int khalf) {
    return ((j < 8) ? 0 : 16) + khalf * 8 + (j & 7);
}

// 16 rows x 16 cols output tile, K = KSTEPS*32.
// A: LDS bf16 row-major [16][lda] (lda multiple of 8).
// W: pre-swizzled bf16 fragments, index ((ct*KSTEPS+ks)*32 + lane)*16 + j.
template <int KSTEPS>
__device__ __forceinline__ v8f wmma_tile(const bf16* __restrict__ A, int lda,
                                         const bf16* __restrict__ W,
                                         int ct, int lane) {
    v8f acc = {};
    const int m  = lane & 15;
    const int kh = lane >> 4;
#pragma unroll
    for (int ks = 0; ks < KSTEPS; ++ks) {
        const bf16* ap = A + m * lda + ks * 32 + kh * 8;
        v8bf alo = *(const v8bf*)ap;          // K = kh*8 .. kh*8+7
        v8bf ahi = *(const v8bf*)(ap + 16);   // K = 16+kh*8 .. 16+kh*8+7
        v16bf a = __builtin_shufflevector(alo, ahi, 0, 1, 2, 3, 4, 5, 6, 7,
                                          8, 9, 10, 11, 12, 13, 14, 15);
        const bf16* wp = W + ((size_t)(ct * KSTEPS + ks) * 32 + lane) * 16;
        v8bf blo = *(const v8bf*)wp;
        v8bf bhi = *(const v8bf*)(wp + 8);
        v16bf b = __builtin_shufflevector(blo, bhi, 0, 1, 2, 3, 4, 5, 6, 7,
                                          8, 9, 10, 11, 12, 13, 14, 15);
        acc = __builtin_amdgcn_wmma_f32_16x16x32_bf16(
            false, a, false, b, (short)0, acc, false, false);
    }
    return acc;
}

// bias + relu -> bf16 LDS tile [16][128]
__device__ __forceinline__ void store_relu_bf(v8f acc, const float* __restrict__ bias,
                                              bf16* __restrict__ O, int ct, int lane) {
    const int n  = ct * 16 + (lane & 15);
    const float bv = bias[n];
    const int mo = (lane >> 4) * 8;
#pragma unroll
    for (int r = 0; r < 8; ++r) {
        float v = acc[r] + bv;
        v = v > 0.f ? v : 0.f;
        O[(mo + r) * HID + n] = (bf16)v;
    }
}

// Parallel LayerNorm stats: 16 threads per row, width-16 shuffle reduction.
template <int K>
__device__ __forceinline__ void ln_stats(const float* __restrict__ sIn,
                                         float* __restrict__ sMu,
                                         float* __restrict__ sRs, int tid) {
    const int r = tid >> 4, g = tid & 15;
    float s = 0.f, ss = 0.f;
    for (int c = g; c < K; c += 16) {
        float v = sIn[r * K + c];
        s += v;
        ss += v * v;
    }
#pragma unroll
    for (int off = 8; off; off >>= 1) {
        s  += __shfl_down(s, off, 16);
        ss += __shfl_down(ss, off, 16);
    }
    if (g == 0) {
        float mu  = s * (1.f / K);
        float var = ss * (1.f / K) - mu * mu;
        var = var > 0.f ? var : 0.f;
        sMu[r] = mu;
        sRs[r] = rsqrtf(var + 1e-5f);
    }
}

// Apply LN + convert to bf16 A-tile (vectorized by 4 columns).
template <int K>
__device__ __forceinline__ void ln_apply(const float* __restrict__ sIn,
                                         const float* __restrict__ sMu,
                                         const float* __restrict__ sRs,
                                         const float* __restrict__ lnG,
                                         const float* __restrict__ lnB,
                                         bf16* __restrict__ sA, int tid) {
    constexpr int C4 = K / 4;
    for (int i4 = tid; i4 < 16 * C4; i4 += 256) {
        int r = i4 / C4, c = (i4 - r * C4) * 4;
        float mu = sMu[r], rs = sRs[r];
#pragma unroll
        for (int j = 0; j < 4; ++j)
            sA[r * K + c + j] = (bf16)((sIn[r * K + c + j] - mu) * rs * lnG[c + j] + lnB[c + j]);
    }
}

// ---------------------------------------------------------------------------
// Weight swizzle: f32 [K x 128] row-major -> bf16 WMMA B fragments (zero pad K).
// ---------------------------------------------------------------------------
__global__ void k_swizzle(const float* __restrict__ W, bf16* __restrict__ out,
                          int Ksrc, int ksteps) {
    int t = blockIdx.x * blockDim.x + threadIdx.x;  // one per (ct, ks, lane)
    int total = 8 * ksteps * 32;
    if (t >= total) return;
    int lane = t & 31;
    int ks   = (t >> 5) % ksteps;
    int kh   = lane >> 4;
    int col  = ((t >> 5) / ksteps) * 16 + (lane & 15);
    bf16 tmp[16];
#pragma unroll
    for (int j = 0; j < 16; ++j) {
        int k = ks * 32 + wmma_k_of(j, kh);
        float v = (k < Ksrc) ? W[(size_t)k * HID + col] : 0.f;
        tmp[j] = (bf16)v;
    }
    bf16* o = out + (size_t)t * 16;
    *(v8bf*)o       = *(const v8bf*)tmp;
    *(v8bf*)(o + 8) = *(const v8bf*)(tmp + 8);
}

__global__ void k_zero4(float4* __restrict__ p, size_t n4) {
    size_t i = (size_t)blockIdx.x * blockDim.x + threadIdx.x;
    if (i < n4) p[i] = float4{0.f, 0.f, 0.f, 0.f};
}

// ---------------------------------------------------------------------------
// Node encoder: [pos(16) | embed(16)] -> LN(32) -> 32x128 -> relu -> 128x128
// -> relu -> 128x128.  One block = 16 nodes, 8 waves.
// ---------------------------------------------------------------------------
__global__ void __launch_bounds__(256)
k_node_encoder(const float* __restrict__ pos, const int* __restrict__ types,
               const float* __restrict__ embed,
               const float* __restrict__ lnG, const float* __restrict__ lnB,
               const bf16* __restrict__ W0, const float* __restrict__ b0,
               const bf16* __restrict__ W1, const float* __restrict__ b1,
               const bf16* __restrict__ W2, const float* __restrict__ b2,
               float* __restrict__ nodeF) {
    __shared__ __align__(16) float sIn[16 * 32];
    __shared__ float sMu[16], sRs[16];
    __shared__ int sTy[16];
    __shared__ __align__(16) bf16 sA[16 * 32];
    __shared__ __align__(16) bf16 sB[16 * HID];
    __shared__ __align__(16) bf16 sC[16 * HID];
    const int tid = threadIdx.x, lane = tid & 31, wv = tid >> 5;
    const int row0 = blockIdx.x * 16;

    if (tid < 16) sTy[tid] = types[row0 + tid];
    __syncthreads();
    // stage 16 rows x 8 float4 (pos | embed row)
    for (int i4 = tid; i4 < 16 * 8; i4 += 256) {
        int r = i4 >> 3, c4 = i4 & 7;
        const float* g = (c4 < 4) ? pos + (size_t)(row0 + r) * 16 + c4 * 4
                                  : embed + (size_t)sTy[r] * 16 + (c4 - 4) * 4;
#if HAVE_ASYNC_LDS
        ASYNC_B128(g, &sIn[r * 32 + c4 * 4]);
#else
        *(float4*)&sIn[r * 32 + c4 * 4] = *(const float4*)g;
#endif
    }
#if HAVE_ASYNC_LDS
    __builtin_amdgcn_s_wait_asynccnt(0);
#endif
    __syncthreads();
    ln_stats<32>(sIn, sMu, sRs, tid);
    __syncthreads();
    ln_apply<32>(sIn, sMu, sRs, lnG, lnB, sA, tid);
    __syncthreads();
    v8f h = wmma_tile<1>(sA, 32, W0, wv, lane);
    store_relu_bf(h, b0, sB, wv, lane);
    __syncthreads();
    h = wmma_tile<4>(sB, HID, W1, wv, lane);
    store_relu_bf(h, b1, sC, wv, lane);
    __syncthreads();
    h = wmma_tile<4>(sC, HID, W2, wv, lane);
    {
        const int n = wv * 16 + (lane & 15);
        const float bv = b2[n];
        const int mo = (lane >> 4) * 8;
#pragma unroll
        for (int r = 0; r < 8; ++r)
            nodeF[(size_t)(row0 + mo + r) * HID + n] = h[r] + bv;
    }
}

// ---------------------------------------------------------------------------
// Edge encoder: edge_attr(3) -> LN(3) -> (pad K=32) 3x128 -> relu -> 128x128
// -> relu -> 128x128.  One block = 16 edges.
// ---------------------------------------------------------------------------
__global__ void __launch_bounds__(256)
k_edge_encoder(const float* __restrict__ eattr,
               const float* __restrict__ lnG, const float* __restrict__ lnB,
               const bf16* __restrict__ W0, const float* __restrict__ b0,
               const bf16* __restrict__ W1, const float* __restrict__ b1,
               const bf16* __restrict__ W2, const float* __restrict__ b2,
               float* __restrict__ edgeF) {
    __shared__ float sIn[16 * 3];
    __shared__ float sMu[16], sRs[16];
    __shared__ __align__(16) bf16 sA[16 * 32];
    __shared__ __align__(16) bf16 sB[16 * HID];
    __shared__ __align__(16) bf16 sC[16 * HID];
    const int tid = threadIdx.x, lane = tid & 31, wv = tid >> 5;
    const int row0 = blockIdx.x * 16;

    for (int i = tid; i < 16 * 3; i += 256)
        sIn[i] = eattr[(size_t)row0 * 3 + i];
    __syncthreads();
    if (tid < 16) {
        float mu = (sIn[tid * 3] + sIn[tid * 3 + 1] + sIn[tid * 3 + 2]) * (1.f / 3.f);
        float var = 0.f;
        for (int c = 0; c < 3; ++c) { float d = sIn[tid * 3 + c] - mu; var += d * d; }
        sMu[tid] = mu;
        sRs[tid] = rsqrtf(var * (1.f / 3.f) + 1e-5f);
    }
    __syncthreads();
    for (int i = tid; i < 16 * 32; i += 256) {
        int r = i >> 5, c = i & 31;
        sA[i] = (c < 3) ? (bf16)((sIn[r * 3 + c] - sMu[r]) * sRs[r] * lnG[c] + lnB[c])
                        : (bf16)0.f;
    }
    __syncthreads();
    v8f h = wmma_tile<1>(sA, 32, W0, wv, lane);
    store_relu_bf(h, b0, sB, wv, lane);
    __syncthreads();
    h = wmma_tile<4>(sB, HID, W1, wv, lane);
    store_relu_bf(h, b1, sC, wv, lane);
    __syncthreads();
    h = wmma_tile<4>(sC, HID, W2, wv, lane);
    {
        const int n = wv * 16 + (lane & 15);
        const float bv = b2[n];
        const int mo = (lane >> 4) * 8;
#pragma unroll
        for (int r = 0; r < 8; ++r)
            edgeF[(size_t)(row0 + mo + r) * HID + n] = h[r] + bv;
    }
}

// ---------------------------------------------------------------------------
// Edge message + residual + scatter:
//   in = [node[dst] | node[src] | edge] (384) -> LN -> 384x128 -> relu
//        -> 128x128 -> relu -> 128x128 = msg
//   edge += msg;  aggr[dst] += msg (f32 atomics).
// ---------------------------------------------------------------------------
__global__ void __launch_bounds__(256)
k_edge_msg(const int* __restrict__ ei, const float* __restrict__ nodeF,
           float* __restrict__ edgeF, float* __restrict__ aggr,
           const float* __restrict__ lnG, const float* __restrict__ lnB,
           const bf16* __restrict__ W0, const float* __restrict__ b0,
           const bf16* __restrict__ W1, const float* __restrict__ b1,
           const bf16* __restrict__ W2, const float* __restrict__ b2,
           int E) {
    __shared__ __align__(16) float sIn[16 * 384];
    __shared__ float sMu[16], sRs[16];
    __shared__ __align__(16) bf16 sA[16 * 384];
    __shared__ __align__(16) bf16 sB[16 * HID];
    __shared__ __align__(16) bf16 sC[16 * HID];
    __shared__ int sSrc[16], sDst[16];
    const int tid = threadIdx.x, lane = tid & 31, wv = tid >> 5;
    const int e0 = blockIdx.x * 16;

    if (tid < 16) {
        sSrc[tid] = ei[e0 + tid];       // edge_index[0] = src (x_j)
        sDst[tid] = ei[E + e0 + tid];   // edge_index[1] = dst (x_i)
    }
    __syncthreads();
    // stage 16 rows x 96 float4: [node[dst] | node[src] | edge]
    for (int i4 = tid; i4 < 16 * 96; i4 += 256) {
        int r = i4 / 96, c4 = i4 - r * 96;
        const float* g;
        if (c4 < 32)      g = nodeF + (size_t)sDst[r] * HID + c4 * 4;
        else if (c4 < 64) g = nodeF + (size_t)sSrc[r] * HID + (c4 - 32) * 4;
        else              g = edgeF + (size_t)(e0 + r) * HID + (c4 - 64) * 4;
#if HAVE_ASYNC_LDS
        ASYNC_B128(g, &sIn[r * 384 + c4 * 4]);
#else
        *(float4*)&sIn[r * 384 + c4 * 4] = *(const float4*)g;
#endif
    }
#if HAVE_ASYNC_LDS
    __builtin_amdgcn_s_wait_asynccnt(0);
#endif
    __syncthreads();
    ln_stats<384>(sIn, sMu, sRs, tid);
    __syncthreads();
    ln_apply<384>(sIn, sMu, sRs, lnG, lnB, sA, tid);
    __syncthreads();
    v8f h = wmma_tile<12>(sA, 384, W0, wv, lane);
    store_relu_bf(h, b0, sB, wv, lane);
    __syncthreads();
    h = wmma_tile<4>(sB, HID, W1, wv, lane);
    store_relu_bf(h, b1, sC, wv, lane);
    __syncthreads();
    h = wmma_tile<4>(sC, HID, W2, wv, lane);
    {
        const int n = wv * 16 + (lane & 15);
        const float bv = b2[n];
        const int mo = (lane >> 4) * 8;
#pragma unroll
        for (int r = 0; r < 8; ++r) {
            int m = mo + r;
            float msg = h[r] + bv;
            edgeF[(size_t)(e0 + m) * HID + n] = sIn[m * 384 + 256 + n] + msg;
            atomicAdd(&aggr[(size_t)sDst[m] * HID + n], msg);
        }
    }
}

// ---------------------------------------------------------------------------
// Node update: in = [node | aggr] (256) -> LN -> 256x128 -> relu -> 128x128
// -> relu -> 128x128;  node += out.
// ---------------------------------------------------------------------------
__global__ void __launch_bounds__(256)
k_node_upd(const float* __restrict__ aggr, float* __restrict__ nodeF,
           const float* __restrict__ lnG, const float* __restrict__ lnB,
           const bf16* __restrict__ W0, const float* __restrict__ b0,
           const bf16* __restrict__ W1, const float* __restrict__ b1,
           const bf16* __restrict__ W2, const float* __restrict__ b2) {
    __shared__ __align__(16) float sIn[16 * 256];
    __shared__ float sMu[16], sRs[16];
    __shared__ __align__(16) bf16 sA[16 * 256];
    __shared__ __align__(16) bf16 sB[16 * HID];
    __shared__ __align__(16) bf16 sC[16 * HID];
    const int tid = threadIdx.x, lane = tid & 31, wv = tid >> 5;
    const int row0 = blockIdx.x * 16;

    // stage 16 rows x 64 float4: [node | aggr]
    for (int i4 = tid; i4 < 16 * 64; i4 += 256) {
        int r = i4 >> 6, c4 = i4 & 63;
        const float* g = (c4 < 32) ? nodeF + (size_t)(row0 + r) * HID + c4 * 4
                                   : aggr + (size_t)(row0 + r) * HID + (c4 - 32) * 4;
#if HAVE_ASYNC_LDS
        ASYNC_B128(g, &sIn[r * 256 + c4 * 4]);
#else
        *(float4*)&sIn[r * 256 + c4 * 4] = *(const float4*)g;
#endif
    }
#if HAVE_ASYNC_LDS
    __builtin_amdgcn_s_wait_asynccnt(0);
#endif
    __syncthreads();
    ln_stats<256>(sIn, sMu, sRs, tid);
    __syncthreads();
    ln_apply<256>(sIn, sMu, sRs, lnG, lnB, sA, tid);
    __syncthreads();
    v8f h = wmma_tile<8>(sA, 256, W0, wv, lane);
    store_relu_bf(h, b0, sB, wv, lane);
    __syncthreads();
    h = wmma_tile<4>(sB, HID, W1, wv, lane);
    store_relu_bf(h, b1, sC, wv, lane);
    __syncthreads();
    h = wmma_tile<4>(sC, HID, W2, wv, lane);
    {
        const int n = wv * 16 + (lane & 15);
        const float bv = b2[n];
        const int mo = (lane >> 4) * 8;
#pragma unroll
        for (int r = 0; r < 8; ++r) {
            int m = mo + r;
            nodeF[(size_t)(row0 + m) * HID + n] = sIn[m * 256 + n] + (h[r] + bv);
        }
    }
}

// ---------------------------------------------------------------------------
// Decoder: node -> 128x128 -> relu -> 128x128 -> relu -> 128x2 (scalar tail).
// ---------------------------------------------------------------------------
__global__ void __launch_bounds__(256)
k_decoder(const float* __restrict__ nodeF,
          const bf16* __restrict__ W0, const float* __restrict__ b0,
          const bf16* __restrict__ W1, const float* __restrict__ b1,
          const float* __restrict__ W2, const float* __restrict__ b2,
          float* __restrict__ out) {
    __shared__ __align__(16) bf16 sA[16 * HID];
    __shared__ __align__(16) bf16 sB[16 * HID];
    __shared__ __align__(16) bf16 sC[16 * HID];
    const int tid = threadIdx.x, lane = tid & 31, wv = tid >> 5;
    const int row0 = blockIdx.x * 16;

    for (int i4 = tid; i4 < 16 * 32; i4 += 256) {
        float4 v = *(const float4*)&nodeF[(size_t)row0 * HID + i4 * 4];
        bf16* o = &sA[i4 * 4];
        o[0] = (bf16)v.x; o[1] = (bf16)v.y; o[2] = (bf16)v.z; o[3] = (bf16)v.w;
    }
    __syncthreads();
    v8f h = wmma_tile<4>(sA, HID, W0, wv, lane);
    store_relu_bf(h, b0, sB, wv, lane);
    __syncthreads();
    h = wmma_tile<4>(sB, HID, W1, wv, lane);
    store_relu_bf(h, b1, sC, wv, lane);
    __syncthreads();
    if (tid < 32) {
        int r = tid >> 1, c = tid & 1;
        float acc = b2[c];
        for (int k = 0; k < HID; ++k)
            acc += (float)sC[r * HID + k] * W2[k * 2 + c];
        out[(size_t)(row0 + r) * 2 + c] = acc;
    }
}

// ---------------------------------------------------------------------------
// Host launcher
// ---------------------------------------------------------------------------
extern "C" void kernel_launch(void* const* d_in, const int* in_sizes, int n_in,
                              void* d_out, int out_size, void* d_ws, size_t ws_size,
                              hipStream_t stream) {
    (void)in_sizes; (void)n_in; (void)out_size; (void)ws_size;
    const int N = 100000, E = 600000;

    const int*   x_types  = (const int*)  d_in[0];
    const float* pos      = (const float*)d_in[1];
    const int*   edge_idx = (const int*)  d_in[2];
    const float* eattr    = (const float*)d_in[3];

    // params flattened in JAX pytree order (dict keys sorted alphabetically):
    // decoder, edge_enc, embed, layers[0..9]{edge,node}, node_enc.
    // Each MLP: "lins" (W0,b0,W1,b1,W2,b2) then "ln" (gamma,beta) if present.
    int p = 4;
    auto F = [&](int i) { return (const float*)d_in[i]; };
    const float *dW0 = F(p), *db0 = F(p + 1), *dW1 = F(p + 2), *db1 = F(p + 3),
                *dW2 = F(p + 4), *db2 = F(p + 5);
    p += 6;
    const float *eeW0 = F(p), *eeb0 = F(p + 1), *eeW1 = F(p + 2), *eeb1 = F(p + 3),
                *eeW2 = F(p + 4), *eeb2 = F(p + 5), *eeG = F(p + 6), *eeBt = F(p + 7);
    p += 8;
    const float* embed = F(p);
    p += 1;
    struct MlpP { const float *W0, *b0, *W1, *b1, *W2, *b2, *G, *B; };
    MlpP Le[10], Ln[10];
    for (int i = 0; i < 10; ++i) {
        Le[i] = { F(p), F(p + 1), F(p + 2), F(p + 3), F(p + 4), F(p + 5), F(p + 6), F(p + 7) };
        p += 8;
        Ln[i] = { F(p), F(p + 1), F(p + 2), F(p + 3), F(p + 4), F(p + 5), F(p + 6), F(p + 7) };
        p += 8;
    }
    const float *neW0 = F(p), *neb0 = F(p + 1), *neW1 = F(p + 2), *neb1 = F(p + 3),
                *neW2 = F(p + 4), *neb2 = F(p + 5), *neG = F(p + 6), *neB = F(p + 7);

    // workspace layout
    char* ws = (char*)d_ws;
    size_t off = 0;
    auto take = [&](size_t bytes) -> void* {
        void* r = ws + off;
        off = (off + bytes + 255) & ~(size_t)255;
        return r;
    };
    float* nodeF = (float*)take((size_t)N * HID * 4);
    float* edgeF = (float*)take((size_t)E * HID * 4);
    float* aggr  = (float*)take((size_t)N * HID * 4);
    auto takeW = [&](int ksteps) -> bf16* { return (bf16*)take((size_t)ksteps * 4096 * 2); };
    bf16 *neW0s = takeW(1), *neW1s = takeW(4), *neW2s = takeW(4);
    bf16 *eeW0s = takeW(1), *eeW1s = takeW(4), *eeW2s = takeW(4);
    bf16 *LeW0s[10], *LeW1s[10], *LeW2s[10], *LnW0s[10], *LnW1s[10], *LnW2s[10];
    for (int i = 0; i < 10; ++i) {
        LeW0s[i] = takeW(12); LeW1s[i] = takeW(4); LeW2s[i] = takeW(4);
        LnW0s[i] = takeW(8);  LnW1s[i] = takeW(4); LnW2s[i] = takeW(4);
    }
    bf16 *dW0s = takeW(4), *dW1s = takeW(4);

    auto swz = [&](const float* W, bf16* dst, int Ksrc, int ksteps) {
        int total = 8 * ksteps * 32;
        k_swizzle<<<(total + 255) / 256, 256, 0, stream>>>(W, dst, Ksrc, ksteps);
    };
    swz(neW0, neW0s, 32, 1);  swz(neW1, neW1s, 128, 4); swz(neW2, neW2s, 128, 4);
    swz(eeW0, eeW0s, 3, 1);   swz(eeW1, eeW1s, 128, 4); swz(eeW2, eeW2s, 128, 4);
    for (int i = 0; i < 10; ++i) {
        swz(Le[i].W0, LeW0s[i], 384, 12);
        swz(Le[i].W1, LeW1s[i], 128, 4);
        swz(Le[i].W2, LeW2s[i], 128, 4);
        swz(Ln[i].W0, LnW0s[i], 256, 8);
        swz(Ln[i].W1, LnW1s[i], 128, 4);
        swz(Ln[i].W2, LnW2s[i], 128, 4);
    }
    swz(dW0, dW0s, 128, 4);
    swz(dW1, dW1s, 128, 4);

    k_node_encoder<<<N / 16, 256, 0, stream>>>(pos, x_types, embed, neG, neB,
                                               neW0s, neb0, neW1s, neb1, neW2s, neb2, nodeF);
    k_edge_encoder<<<E / 16, 256, 0, stream>>>(eattr, eeG, eeBt,
                                               eeW0s, eeb0, eeW1s, eeb1, eeW2s, eeb2, edgeF);
    for (int l = 0; l < 10; ++l) {
        size_t n4 = (size_t)N * HID / 4;
        k_zero4<<<(unsigned)((n4 + 255) / 256), 256, 0, stream>>>((float4*)aggr, n4);
        k_edge_msg<<<E / 16, 256, 0, stream>>>(edge_idx, nodeF, edgeF, aggr,
                                               Le[l].G, Le[l].B,
                                               LeW0s[l], Le[l].b0, LeW1s[l], Le[l].b1,
                                               LeW2s[l], Le[l].b2, E);
        k_node_upd<<<N / 16, 256, 0, stream>>>(aggr, nodeF,
                                               Ln[l].G, Ln[l].B,
                                               LnW0s[l], Ln[l].b0, LnW1s[l], Ln[l].b1,
                                               LnW2s[l], Ln[l].b2);
    }
    k_decoder<<<N / 16, 256, 0, stream>>>(nodeF, dW0s, db0, dW1s, db1, dW2, db2,
                                          (float*)d_out);
}
